// STSelfAttention_73770358276500
// MI455X (gfx1250) — matrix operand order repaired and checked by reading, not compiled, not run on hardware
//
#include <hip/hip_runtime.h>
#include <cmath>

#define B_ 8
#define T_ 12
#define N_ 512
#define M_ (B_*T_*N_)          // 49152 rows
#define EPROJ 192              // tq16 tk16 tv16 | gq32 gk32 gv32 | sq16 sk16 sv16
#define SCALE_ 0.35355339059327373f   // 8^-0.5

typedef __attribute__((ext_vector_type(16))) _Float16 v16h;
typedef __attribute__((ext_vector_type(8)))  float    v8f;

// ---------------- WMMA helpers (CDNA5 gfx1250, wave32) ----------------

__device__ __forceinline__ v8f wmma16(v16h a, v16h b, v8f c) {
  // D = A(16x32 f16) x B(32x16 f16) + C(16x16 f32)
  return __builtin_amdgcn_wmma_f32_16x16x32_f16(false, a, false, b, (short)0, c,
                                                false, false);
}

// A fragment (16x32): lane m<16 rows; K contiguous per lane half (k = 16*g + i).
__device__ __forceinline__ v16h frag_a(const _Float16* s, int ld, int k0) {
  int l = threadIdx.x & 31, m = l & 15, g = (l >> 4) & 1;
  v16h a;
#pragma unroll
  for (int i = 0; i < 16; ++i) {
    int k = (g << 4) + i;
    a[i] = s[m * ld + k0 + k];
  }
  return a;
}

__device__ __forceinline__ v16h frag_a_pad(const _Float16* s, int ld, int kmax) {
  int l = threadIdx.x & 31, m = l & 15, g = (l >> 4) & 1;
  v16h a;
#pragma unroll
  for (int i = 0; i < 16; ++i) {
    int k = (g << 4) + i;
    a[i] = (k < kmax) ? s[m * ld + k] : (_Float16)0.f;
  }
  return a;
}

// B fragment (32x16), memory tile N-major: B[k][n] = s[n*ld + k0 + k]
__device__ __forceinline__ v16h frag_b_nk(const _Float16* s, int ld, int k0) {
  int l = threadIdx.x & 31, n = l & 15, kg = (l >> 4) & 1;
  v16h b;
#pragma unroll
  for (int i = 0; i < 16; ++i) {
    int k = (kg << 4) + i;
    b[i] = s[n * ld + k0 + k];
  }
  return b;
}

__device__ __forceinline__ v16h frag_b_nk_pad(const _Float16* s, int ld, int kmax) {
  int l = threadIdx.x & 31, n = l & 15, kg = (l >> 4) & 1;
  v16h b;
#pragma unroll
  for (int i = 0; i < 16; ++i) {
    int k = (kg << 4) + i;
    b[i] = (k < kmax) ? s[n * ld + k] : (_Float16)0.f;
  }
  return b;
}

// B fragment from K-major tile: B[k][n] = s[k*ld + n] (zero-padded)
__device__ __forceinline__ v16h frag_b_kn_pad(const _Float16* s, int ld,
                                              int kmax, int nmax) {
  int l = threadIdx.x & 31, n = l & 15, kg = (l >> 4) & 1;
  v16h b;
#pragma unroll
  for (int i = 0; i < 16; ++i) {
    int k = (kg << 4) + i;
    b[i] = (k < kmax && n < nmax) ? s[k * ld + n] : (_Float16)0.f;
  }
  return b;
}

// ---------------- small f32->f16 conversion (weights only) ----------------

__global__ void k_cvt(const float* __restrict__ src, _Float16* __restrict__ dst, int n) {
  int i = blockIdx.x * blockDim.x + threadIdx.x;
  if (i < n) dst[i] = (_Float16)src[i];
}

// ---------------- fused projection GEMM: Y[m][0..191] = X_f32[m][:64] . W^T ----
// One wave computes a 16(M) x 64(E) strip: A fragments loaded once, reused for
// 4 N-tiles (8 WMMAs). f32->f16 conversion fused into LDS staging; output
// bounced through LDS for coalesced b128 stores.

__global__ void k_proj(const float* __restrict__ X, const _Float16* __restrict__ W,
                       _Float16* __restrict__ Y) {
  __shared__ _Float16 sa[16 * 64];   // A tile (f16, converted in staging)
  __shared__ _Float16 sb[64 * 64];   // 64 weight rows
  __shared__ _Float16 so[16 * 64];   // output staging
  int m0 = blockIdx.x * 16, e0 = blockIdx.y * 64;
  int l = threadIdx.x, n = l & 15, lg = l >> 4;
  {
    int row = l >> 1, c0 = (l & 1) * 32;          // 32 floats per lane
    const float4* gx = (const float4*)(X + (size_t)(m0 + row) * 64 + c0);
#pragma unroll
    for (int j = 0; j < 8; ++j) {
      float4 v = gx[j];
      _Float16* d = sa + row * 64 + c0 + j * 4;
      d[0] = (_Float16)v.x; d[1] = (_Float16)v.y;
      d[2] = (_Float16)v.z; d[3] = (_Float16)v.w;
    }
  }
  {
#pragma unroll
    for (int j = 0; j < 16; ++j) {                // 512 b128 chunks / 32 lanes
      int idx = l * 16 + j;
      int row = idx >> 3, c0 = (idx & 7) * 8;
      *(uint4*)(sb + row * 64 + c0) =
          *(const uint4*)(W + (size_t)(e0 + row) * 64 + c0);
    }
  }
  __syncthreads();
  v16h a0 = frag_a(sa, 64, 0);
  v16h a1 = frag_a(sa, 64, 32);
#pragma unroll
  for (int et = 0; et < 4; ++et) {
    v8f c = {};
    c = wmma16(a0, frag_b_nk(sb + et * 16 * 64, 64, 0),  c);
    c = wmma16(a1, frag_b_nk(sb + et * 16 * 64, 64, 32), c);
#pragma unroll
    for (int i = 0; i < 8; ++i)
      so[(i + lg * 8) * 64 + et * 16 + n] = (_Float16)c[i];
  }
  __syncthreads();
  {
    int row = l >> 1, c0 = (l & 1) * 32;
    _Float16* dst = Y + (size_t)(m0 + row) * EPROJ + e0 + c0;
    const uint4* s = (const uint4*)(so + row * 64 + c0);
    uint4* d = (uint4*)dst;
    d[0] = s[0]; d[1] = s[1]; d[2] = s[2]; d[3] = s[3];
  }
}

// ---------------- final GEMM (E=64) with bias, coalesced fp32 out ----------------

__global__ void k_gemm_final(const _Float16* __restrict__ X, const _Float16* __restrict__ W,
                             const float* __restrict__ bias, float* __restrict__ Y) {
  __shared__ _Float16 sa[16 * 64];
  __shared__ _Float16 sb[64 * 64];
  __shared__ float    so[16 * 64];
  int m0 = blockIdx.x * 16;
  int l = threadIdx.x, n = l & 15, lg = l >> 4;
  {
    int row = l >> 1, c0 = (l & 1) * 32;
    const uint4* gx = (const uint4*)(X + (size_t)(m0 + row) * 64 + c0);
    uint4* dx = (uint4*)(sa + row * 64 + c0);
    dx[0] = gx[0]; dx[1] = gx[1]; dx[2] = gx[2]; dx[3] = gx[3];
  }
  {
#pragma unroll
    for (int j = 0; j < 16; ++j) {
      int idx = l * 16 + j;
      int row = idx >> 3, c0 = (idx & 7) * 8;
      *(uint4*)(sb + row * 64 + c0) =
          *(const uint4*)(W + (size_t)row * 64 + c0);
    }
  }
  __syncthreads();
  v16h a0 = frag_a(sa, 64, 0);
  v16h a1 = frag_a(sa, 64, 32);
#pragma unroll
  for (int et = 0; et < 4; ++et) {
    v8f c = {};
    c = wmma16(a0, frag_b_nk(sb + et * 16 * 64, 64, 0),  c);
    c = wmma16(a1, frag_b_nk(sb + et * 16 * 64, 64, 32), c);
    float bb = bias[et * 16 + n];
#pragma unroll
    for (int i = 0; i < 8; ++i)
      so[(i + lg * 8) * 64 + et * 16 + n] = c[i] + bb;
  }
  __syncthreads();
  {
    int row = l >> 1, c0 = (l & 1) * 32;       // 32 floats per lane
    float* dst = Y + (size_t)(m0 + row) * 64 + c0;
    const uint4* s = (const uint4*)(so + row * 64 + c0);
    uint4* d = (uint4*)dst;
#pragma unroll
    for (int j = 0; j < 8; ++j) d[j] = s[j];
  }
}

// ---------------- pattern keys/values (tiny: 96 x 16 x 32, K=64) ----------------

__global__ void k_patkv(const float* __restrict__ pkeys,
                        const float* __restrict__ kw, const float* __restrict__ kb,
                        const float* __restrict__ vw, const float* __restrict__ vb,
                        _Float16* __restrict__ pk, _Float16* __restrict__ pv) {
  int bt = blockIdx.x;
  int e = threadIdx.x;
  int p = e >> 5, c = e & 31;
  const float* src = pkeys + ((size_t)bt * 16 + p) * 64;
  float aK = kb[c], aV = vb[c];
  for (int k = 0; k < 64; ++k) {
    float xv = src[k];
    aK += xv * kw[c * 64 + k];
    aV += xv * vw[c * 64 + k];
  }
  pk[((size_t)bt * 16 + p) * 32 + c] = (_Float16)aK;
  pv[((size_t)bt * 16 + p) * 32 + c] = (_Float16)aV;
}

// ---------------- pattern attention: geo_k += softmax(pq.pk^T * s) . pv ----------------
// x_patterns f32->f16 conversion fused into staging (no xph buffer).

__global__ void k_pattern(const float* __restrict__ xp,
                          const _Float16* __restrict__ patqh,
                          const float* __restrict__ patqb,
                          const _Float16* __restrict__ pk,
                          const _Float16* __restrict__ pv,
                          _Float16* __restrict__ projO) {
  __shared__ _Float16 sx[16 * 64];
  __shared__ _Float16 swq[32 * 64];
  __shared__ _Float16 spk[16 * 32];
  __shared__ _Float16 spv[16 * 32];
  __shared__ _Float16 spq[16 * 32];
  __shared__ float    ss[16 * 16];
  __shared__ _Float16 spa[16 * 16];
  int bt = blockIdx.x, m0 = blockIdx.y * 16;
  int l = threadIdx.x, n = l & 15, lg = l >> 4;
  {
    int row = l >> 1, c0 = (l & 1) * 32;
    const float4* g = (const float4*)(xp + ((size_t)bt * 512 + m0 + row) * 64 + c0);
#pragma unroll
    for (int j = 0; j < 8; ++j) {
      float4 v = g[j];
      _Float16* d = sx + row * 64 + c0 + j * 4;
      d[0] = (_Float16)v.x; d[1] = (_Float16)v.y;
      d[2] = (_Float16)v.z; d[3] = (_Float16)v.w;
    }
  }
  {
    const uint4* g = (const uint4*)(patqh + (size_t)l * 64);
    uint4* d = (uint4*)(swq + l * 64);
#pragma unroll
    for (int j = 0; j < 8; ++j) d[j] = g[j];
  }
  {
    int row = l >> 1, c0 = (l & 1) * 16;
    const uint4* gk = (const uint4*)(pk + ((size_t)bt * 16 + row) * 32 + c0);
    uint4* dk = (uint4*)(spk + row * 32 + c0);
    dk[0] = gk[0]; dk[1] = gk[1];
    const uint4* gv = (const uint4*)(pv + ((size_t)bt * 16 + row) * 32 + c0);
    uint4* dv = (uint4*)(spv + row * 32 + c0);
    dv[0] = gv[0]; dv[1] = gv[1];
  }
  __syncthreads();
  v8f zero = {};
  v16h a0 = frag_a(sx, 64, 0);
  v16h a1 = frag_a(sx, 64, 32);
  // pq = x_patterns_tile @ pat_q_w^T + b   (two 16-col tiles, K=64)
#pragma unroll
  for (int ct = 0; ct < 2; ++ct) {
    v8f c = zero;
    c = wmma16(a0, frag_b_nk(swq + ct * 16 * 64, 64, 0),  c);
    c = wmma16(a1, frag_b_nk(swq + ct * 16 * 64, 64, 32), c);
    float bb = patqb[ct * 16 + n];
#pragma unroll
    for (int i = 0; i < 8; ++i)
      spq[(i + lg * 8) * 32 + ct * 16 + n] = (_Float16)(c[i] + bb);
  }
  __syncthreads();
  // s = pq @ pk^T  (K=32, exact)
  v8f sc = wmma16(frag_a(spq, 32, 0), frag_b_nk(spk, 32, 0), zero);
#pragma unroll
  for (int i = 0; i < 8; ++i) ss[(i + lg * 8) * 16 + n] = sc[i] * SCALE_;
  __syncthreads();
  if (l < 16) {
    float mx = -3.4e38f;
    for (int j = 0; j < 16; ++j) mx = fmaxf(mx, ss[l * 16 + j]);
    float e[16], sum = 0.f;
    for (int j = 0; j < 16; ++j) { e[j] = __expf(ss[l * 16 + j] - mx); sum += e[j]; }
    float inv = 1.f / sum;
    for (int j = 0; j < 16; ++j) spa[l * 16 + j] = (_Float16)(e[j] * inv);
  }
  __syncthreads();
  // geo_k(cols 80..111) += pa @ pv  (K=16 padded to 32)
  v16h pa = frag_a_pad(spa, 16, 16);
#pragma unroll
  for (int ct = 0; ct < 2; ++ct) {
    v8f g = wmma16(pa, frag_b_kn_pad(spv + ct * 16, 32, 16, 16), zero);
#pragma unroll
    for (int i = 0; i < 8; ++i) {
      size_t idx = ((size_t)bt * 512 + m0 + i + lg * 8) * EPROJ + 80 + ct * 16 + n;
      projO[idx] = (_Float16)((float)projO[idx] + g[i]);
    }
  }
}

// ---------------- temporal branch (T=12, tiny -> VALU) ----------------

__global__ void k_temporal(const _Float16* __restrict__ proj, _Float16* __restrict__ cat) {
  int idx = blockIdx.x * blockDim.x + threadIdx.x;
  if (idx >= B_ * N_ * 2 * T_) return;
  int t = idx % T_;
  int h = (idx / T_) & 1;
  int nd = (idx / (T_ * 2)) % N_;
  int b = idx / (T_ * 2 * N_);
  float q[8];
  const _Float16* qp = proj + ((size_t)(b * T_ + t) * N_ + nd) * EPROJ + h * 8;
#pragma unroll
  for (int d = 0; d < 8; ++d) q[d] = (float)qp[d];
  float sc[12], mx = -3.4e38f;
  for (int s = 0; s < T_; ++s) {
    const _Float16* kp = proj + ((size_t)(b * T_ + s) * N_ + nd) * EPROJ + 16 + h * 8;
    float dot = 0.f;
#pragma unroll
    for (int d = 0; d < 8; ++d) dot += q[d] * (float)kp[d];
    sc[s] = dot * SCALE_;
    mx = fmaxf(mx, sc[s]);
  }
  float sum = 0.f, o[8] = {0, 0, 0, 0, 0, 0, 0, 0};
  for (int s = 0; s < T_; ++s) {
    float w = __expf(sc[s] - mx);
    sum += w;
    const _Float16* vp = proj + ((size_t)(b * T_ + s) * N_ + nd) * EPROJ + 32 + h * 8;
#pragma unroll
    for (int d = 0; d < 8; ++d) o[d] += w * (float)vp[d];
  }
  float inv = 1.f / sum;
  _Float16* op = cat + ((size_t)(b * T_ + t) * N_ + nd) * 64 + h * 8;
#pragma unroll
  for (int d = 0; d < 8; ++d) op[d] = (_Float16)(o[d] * inv);
}

// ---------------- flash attention over N=512 (geo 4 heads + sem 2 heads) ---------
// 4 waves per workgroup share one staged K/V tile; each wave owns a 16-row
// query strip (64 query rows per block, blockIdx.z covers N/64 = 8 groups).

__global__ void __launch_bounds__(128)
k_attn(const _Float16* __restrict__ proj,
       const unsigned char* __restrict__ gmask,
       const unsigned char* __restrict__ smask,
       _Float16* __restrict__ cat) {
  __shared__ _Float16 sk[16 * 8];
  __shared__ _Float16 sv[16 * 8];
  __shared__ _Float16 sq[4][16 * 8];
  __shared__ _Float16 sp[4][16 * 16];
  int bt = blockIdx.x, ha = blockIdx.y;
  int w = threadIdx.x >> 5;                 // wave id 0..3
  int rt = blockIdx.z * 4 + w;              // query row-tile
  int qoff, koff, voff, ooff;
  const unsigned char* mask;
  if (ha < 4) { qoff = 48 + ha * 8;  koff = 80 + ha * 8;  voff = 112 + ha * 8;
                ooff = 16 + ha * 8;  mask = gmask; }
  else { int hs = ha - 4;
         qoff = 144 + hs * 8; koff = 160 + hs * 8; voff = 176 + hs * 8;
         ooff = 48 + hs * 8;  mask = smask; }
  int l = threadIdx.x & 31, n = l & 15, lg = l >> 4;
  size_t base = (size_t)bt * 512;
  if (l < 16)
    *(uint4*)(&sq[w][l * 8]) =
        *(const uint4*)(proj + (base + rt * 16 + l) * EPROJ + qoff);
  __syncthreads();
  v16h qa = frag_a_pad(sq[w], 8, 8);        // K padded 8 -> 32
  v8f zero = {};
  v8f oacc = {};
  float mrow[8], lrow[8];
#pragma unroll
  for (int i = 0; i < 8; ++i) { mrow[i] = -__builtin_huge_valf(); lrow[i] = 0.f; }

  for (int kt = 0; kt < 32; ++kt) {
    __syncthreads();
    if (threadIdx.x < 16) {
      *(uint4*)(&sk[threadIdx.x * 8]) =
          *(const uint4*)(proj + (base + kt * 16 + threadIdx.x) * EPROJ + koff);
      *(uint4*)(&sv[threadIdx.x * 8]) =
          *(const uint4*)(proj + (base + kt * 16 + threadIdx.x) * EPROJ + voff);
    }
    if (kt + 1 < 32 && threadIdx.x < 16)  // pull the next K tile toward the caches
      __builtin_prefetch(proj + (base + (kt + 1) * 16 + threadIdx.x) * EPROJ + koff, 0, 1);
    __syncthreads();
    v8f s = wmma16(qa, frag_b_nk_pad(sk, 8, 8), zero);  // scores [16q x 16k]
    int kcol = kt * 16 + n;
    const unsigned char* mcol = mask + kcol;
#pragma unroll
    for (int i = 0; i < 8; ++i) {
      int qrow = rt * 16 + i + lg * 8;
      float sv_ = s[i] * SCALE_;
      if (mcol[(size_t)qrow * 512]) sv_ = -__builtin_huge_valf();
      s[i] = sv_;
    }
#pragma unroll
    for (int i = 0; i < 8; ++i) {
      float v = s[i];                       // row max across 16 column lanes
      v = fmaxf(v, __shfl_xor(v, 1, 32));
      v = fmaxf(v, __shfl_xor(v, 2, 32));
      v = fmaxf(v, __shfl_xor(v, 4, 32));
      v = fmaxf(v, __shfl_xor(v, 8, 32));
      float mn = fmaxf(mrow[i], v);
      float p = (s[i] <= -3.0e38f) ? 0.f : __expf(s[i] - mn);
      float alpha = (mrow[i] <= -3.0e38f) ? 0.f : __expf(mrow[i] - mn);
      float ps = p;
      ps += __shfl_xor(ps, 1, 32);
      ps += __shfl_xor(ps, 2, 32);
      ps += __shfl_xor(ps, 4, 32);
      ps += __shfl_xor(ps, 8, 32);
      lrow[i] = lrow[i] * alpha + ps;
      oacc[i] = oacc[i] * alpha;
      mrow[i] = mn;
      sp[w][(i + lg * 8) * 16 + n] = (_Float16)p;  // C-layout -> LDS -> A-layout
    }
    __syncthreads();
    oacc = wmma16(frag_a_pad(sp[w], 16, 16), frag_b_kn_pad(sv, 8, 16, 8), oacc);
  }
  if (n < 8) {
#pragma unroll
    for (int i = 0; i < 8; ++i) {
      int row = rt * 16 + i + lg * 8;
      cat[(base + row) * 64 + ooff + n] = (_Float16)(oacc[i] / lrow[i]);
    }
  }
}

// ---------------- host-side launch ----------------

extern "C" void kernel_launch(void* const* d_in, const int* in_sizes, int n_in,
                              void* d_out, int out_size, void* d_ws, size_t ws_size,
                              hipStream_t stream) {
  (void)in_sizes; (void)n_in; (void)out_size; (void)ws_size;
  const float* x     = (const float*)d_in[0];
  const float* xp    = (const float*)d_in[1];
  const float* pkeys = (const float*)d_in[2];
  const unsigned char* gmask = (const unsigned char*)d_in[3];
  const unsigned char* smask = (const unsigned char*)d_in[4];
  const float* w_tq = (const float*)d_in[5];
  const float* w_tk = (const float*)d_in[6];
  const float* w_tv = (const float*)d_in[7];
  const float* w_gq = (const float*)d_in[8];
  const float* w_gk = (const float*)d_in[9];
  const float* w_gv = (const float*)d_in[10];
  const float* w_sq = (const float*)d_in[11];
  const float* w_sk = (const float*)d_in[12];
  const float* w_sv = (const float*)d_in[13];
  const float* patqw = (const float*)d_in[14];
  const float* patqb = (const float*)d_in[15];
  const float* patkw = (const float*)d_in[16];
  const float* patkb = (const float*)d_in[17];
  const float* patvw = (const float*)d_in[18];
  const float* patvb = (const float*)d_in[19];
  const float* projw = (const float*)d_in[20];
  const float* projb = (const float*)d_in[21];

  char* ws = (char*)d_ws;
  size_t off = 0;
  auto alloc = [&](size_t bytes) -> void* {
    void* p = ws + off;
    off += (bytes + 255) & ~(size_t)255;
    return p;
  };
  _Float16* Wcat   = (_Float16*)alloc((size_t)EPROJ * 64 * 2);
  _Float16* patqh  = (_Float16*)alloc((size_t)32 * 64 * 2);
  _Float16* wprojh = (_Float16*)alloc((size_t)64 * 64 * 2);
  _Float16* pkh    = (_Float16*)alloc((size_t)96 * 16 * 32 * 2);
  _Float16* pvh    = (_Float16*)alloc((size_t)96 * 16 * 32 * 2);
  _Float16* projO  = (_Float16*)alloc((size_t)M_ * EPROJ * 2);
  _Float16* cat    = (_Float16*)alloc((size_t)M_ * 64 * 2);

  auto gs = [](int n) { return (n + 255) / 256; };
  // pack projection weights into Wcat rows: tq0 tk16 tv32 gq48 gk80 gv112 sq144 sk160 sv176
  k_cvt<<<gs(16 * 64), 256, 0, stream>>>(w_tq, Wcat + 0   * 64, 16 * 64);
  k_cvt<<<gs(16 * 64), 256, 0, stream>>>(w_tk, Wcat + 16  * 64, 16 * 64);
  k_cvt<<<gs(16 * 64), 256, 0, stream>>>(w_tv, Wcat + 32  * 64, 16 * 64);
  k_cvt<<<gs(32 * 64), 256, 0, stream>>>(w_gq, Wcat + 48  * 64, 32 * 64);
  k_cvt<<<gs(32 * 64), 256, 0, stream>>>(w_gk, Wcat + 80  * 64, 32 * 64);
  k_cvt<<<gs(32 * 64), 256, 0, stream>>>(w_gv, Wcat + 112 * 64, 32 * 64);
  k_cvt<<<gs(16 * 64), 256, 0, stream>>>(w_sq, Wcat + 144 * 64, 16 * 64);
  k_cvt<<<gs(16 * 64), 256, 0, stream>>>(w_sk, Wcat + 160 * 64, 16 * 64);
  k_cvt<<<gs(16 * 64), 256, 0, stream>>>(w_sv, Wcat + 176 * 64, 16 * 64);
  k_cvt<<<gs(32 * 64), 256, 0, stream>>>(patqw, patqh,  32 * 64);
  k_cvt<<<gs(64 * 64), 256, 0, stream>>>(projw, wprojh, 64 * 64);

  k_patkv<<<96, 512, 0, stream>>>(pkeys, patkw, patkb, patvw, patvb, pkh, pvh);
  k_proj<<<dim3(M_ / 16, 3), 32, 0, stream>>>(x, Wcat, projO);
  k_pattern<<<dim3(96, 32), 32, 0, stream>>>(xp, patqh, patqb, pkh, pvh, projO);
  k_temporal<<<gs(B_ * N_ * 2 * T_), 256, 0, stream>>>(projO, cat);
  k_attn<<<dim3(96, 6, 8), 128, 0, stream>>>(projO, gmask, smask, cat);
  k_gemm_final<<<M_ / 16, 32, 0, stream>>>(cat, wprojh, projb, (float*)d_out);
}
